// FrequencyEncoder_15985868276364
// MI455X (gfx1250) — compile-verified, matrix operand order checked by reading, not run on hardware
//
#include <hip/hip_runtime.h>

typedef __attribute__((ext_vector_type(16))) _Float16 v16h;
typedef __attribute__((ext_vector_type(8)))  _Float16 v8h;
typedef __attribute__((ext_vector_type(8)))  float    v8f;

// CDNA5 16-bit A/B fragment K-map (wave32):
//   reg v (0..7), elements {2v,2v+1} <-> K = ((v&4)?16:0) + half*8 + (v&3)*2 (+0,+1)
//   => frag elements 0..7  = K (half*8 + 0..7),  elements 8..15 = K (16 + half*8 + 0..7)
//   => a B fragment is two contiguous 8-half (16B) chunks of a K-major activation row.

#define XPAD 280   // halves; 560B row: 16B-aligned, stride 140 dw, lanes*140 -> distinct banks
#define APAD 196   // f32;   784B row: 16B-aligned, stride 196 dw %64 = 4 -> distinct banks

// --- prepack f32 weights into f16 WMMA-A-fragment order: wp[tile][kb][lane][e] ---
__global__ void pack_weights(const float* __restrict__ w, _Float16* __restrict__ wp, int ntiles) {
    int id = blockIdx.x * 256 + threadIdx.x;
    if (id >= ntiles * 8 * 32 * 16) return;
    int e = id & 15, lane = (id >> 4) & 31, kb = (id >> 9) & 7, tile = id >> 12;
    int m = tile * 16 + (lane & 15);
    int half = lane >> 4;
    int v = e >> 1;
    int kk = kb * 32 + ((v & 4) ? 16 : 0) + half * 8 + (v & 3) * 2 + (e & 1);
    wp[id] = (_Float16)w[m * 256 + kk];
}

__device__ __forceinline__ v16h load_bfrag(const _Float16* row, int kbase, int half) {
    const _Float16* base = row + kbase + half * 8;
    v8h lo = *(const v8h*)(base);        // K = kbase+half*8 .. +7
    v8h hi = *(const v8h*)(base + 16);   // K = kbase+16+half*8 .. +7
    return __builtin_shufflevector(lo, hi, 0,1,2,3,4,5,6,7,8,9,10,11,12,13,14,15);
}

template <bool PACKED>
__launch_bounds__(256)
__global__ void freq_encoder_kernel(const float* __restrict__ img,
                                    const float* __restrict__ bits,
                                    const float* __restrict__ w1,
                                    const float* __restrict__ b1,
                                    const float* __restrict__ w2,
                                    const float* __restrict__ b2,
                                    const _Float16* __restrict__ w1p,
                                    const _Float16* __restrict__ w2p,
                                    float* __restrict__ out)
{
    __shared__ float    Dm[8][8];        // orthonormal DCT matrix
    __shared__ float    Xb[48][64];      // 48 input blocks (3 ch x 16 pos)
    __shared__ float    Tb[48][64];      // stage temp (DCT + IDCT)
    __shared__ _Float16 xT[16][XPAD];    // layer-1 B matrix [pos][cin]
    __shared__ _Float16 hT[16][XPAD];    // layer-2 B matrix [pos][mid]
    __shared__ float    ars[16][APAD];   // layer-2 out     [pos][cout]

    const int tid = threadIdx.x;
    const int bid = blockIdx.x;
    const int b   = bid >> 8;
    const int rr  = bid & 255;
    const int ih  = rr >> 2;             // block row 0..63
    const int iw0 = (rr & 3) << 4;       // first block col (0,16,32,48)

    if (tid < 64) {
        int k = tid >> 3, m = tid & 7;
        float v = (k == 0) ? 0.35355339059327373f
                           : 0.5f * cosf(0.19634954084936207f * (float)((2 * m + 1) * k));
        Dm[k][m] = v;
    }

    // load image tile: 3 ch x 8 rows x 128 cols, float4-coalesced
    {
        const size_t imgbase = (size_t)b * 3u * 512u * 512u;
        const int rrow = tid >> 5;
        const int seg  = tid & 31;
        const int p    = seg >> 1;
        const int c0   = (seg & 1) * 4;
        for (int ccol = 0; ccol < 3; ++ccol) {
            const float4 v = *(const float4*)(img + imgbase
                               + ((size_t)ccol * 512u + (size_t)(ih * 8 + rrow)) * 512u
                               + (size_t)(iw0 * 8 + seg * 4));
            float* dst = &Xb[ccol * 16 + p][rrow * 8 + c0];
            dst[0] = v.x; dst[1] = v.y; dst[2] = v.z; dst[3] = v.w;
        }
    }

    // bits -> channels 192..255 (broadcast over positions)
    for (int s = 0; s < 4; ++s) {
        int idx = tid + s * 256;
        int p = idx >> 6, q = idx & 63;
        xT[p][192 + q] = (_Float16)bits[b * 64 + q];
    }
    __syncthreads();

    // DCT stage 1: T[k][j] = sum_m X[k][m] * D[j][m]
    for (int s = 0; s < 12; ++s) {
        int o = tid + s * 256;
        int blk = o >> 6, k = (o >> 3) & 7, j = o & 7;
        float acc = 0.f;
#pragma unroll
        for (int m = 0; m < 8; ++m) acc += Xb[blk][k * 8 + m] * Dm[j][m];
        Tb[blk][k * 8 + j] = acc;
    }
    __syncthreads();
    // DCT stage 2: Y[i][j] = sum_k D[i][k] * T[k][j]  -> f16 xT[pos][color*64+i*8+j]
    for (int s = 0; s < 12; ++s) {
        int o = tid + s * 256;
        int blk = o >> 6, i = (o >> 3) & 7, j = o & 7;
        float acc = 0.f;
#pragma unroll
        for (int k = 0; k < 8; ++k) acc += Dm[i][k] * Tb[blk][k * 8 + j];
        int p = blk & 15, color = blk >> 4;
        xT[p][color * 64 + i * 8 + j] = (_Float16)acc;
    }
    __syncthreads();

    const int wv   = tid >> 5;
    const int lane = tid & 31;
    const int lm   = lane & 15;
    const int half = lane >> 4;
    const v16h* w1f = (const v16h*)w1p;  // [tile][kb][lane]
    const v16h* w2f = (const v16h*)w2p;

    // --- Layer 1: mid = relu(W1 @ x + b1); wave wv owns M-tiles 2*wv, 2*wv+1 ---
    {
        v8f acc0 = {}; v8f acc1 = {};
        const int m0 = wv * 32 + lm;
        for (int kb = 0; kb < 8; ++kb) {
            const int kbase = kb * 32;
            v16h bf = load_bfrag(&xT[lm][0], kbase, half);
            v16h a0, a1;
            if (PACKED) {
                a0 = w1f[((2 * wv) * 8 + kb) * 32 + lane];
                a1 = w1f[((2 * wv + 1) * 8 + kb) * 32 + lane];
            } else {
#pragma unroll
                for (int v = 0; v < 8; ++v) {
                    const int kk = kbase + ((v & 4) ? 16 : 0) + half * 8 + (v & 3) * 2;
                    const float* p0 = w1 + m0 * 256 + kk;
                    const float* p1 = w1 + (m0 + 16) * 256 + kk;
                    a0[2 * v] = (_Float16)p0[0]; a0[2 * v + 1] = (_Float16)p0[1];
                    a1[2 * v] = (_Float16)p1[0]; a1[2 * v + 1] = (_Float16)p1[1];
                }
            }
            acc0 = __builtin_amdgcn_wmma_f32_16x16x32_f16(false, a0, false, bf, (short)0, acc0, false, false);
            acc1 = __builtin_amdgcn_wmma_f32_16x16x32_f16(false, a1, false, bf, (short)0, acc1, false, false);
        }
        // D layout: reg r -> M = r + 8*half; rows are consecutive -> one b128 store per tile
        const int mb0 = wv * 32 + half * 8;
        const int mb1 = mb0 + 16;
        v8h h0, h1;
#pragma unroll
        for (int r = 0; r < 8; ++r) {
            float v0 = acc0[r] + b1[mb0 + r];
            float v1 = acc1[r] + b1[mb1 + r];
            h0[r] = (_Float16)(v0 > 0.f ? v0 : 0.f);
            h1[r] = (_Float16)(v1 > 0.f ? v1 : 0.f);
        }
        *(v8h*)&hT[lm][mb0] = h0;
        *(v8h*)&hT[lm][mb1] = h1;
    }
    __syncthreads();

    // --- Layer 2: ars = W2 @ mid + b2; 12 tiles round-robin over 8 waves ---
    for (int t = wv; t < 12; t += 8) {    // wave-uniform trip count -> EXEC stays full
        v8f acc = {};
        const int m0 = t * 16 + lm;
        for (int kb = 0; kb < 8; ++kb) {
            const int kbase = kb * 32;
            v16h bf = load_bfrag(&hT[lm][0], kbase, half);
            v16h a;
            if (PACKED) {
                a = w2f[(t * 8 + kb) * 32 + lane];
            } else {
#pragma unroll
                for (int v = 0; v < 8; ++v) {
                    const int kk = kbase + ((v & 4) ? 16 : 0) + half * 8 + (v & 3) * 2;
                    const float* p0 = w2 + m0 * 256 + kk;
                    a[2 * v] = (_Float16)p0[0]; a[2 * v + 1] = (_Float16)p0[1];
                }
            }
            acc = __builtin_amdgcn_wmma_f32_16x16x32_f16(false, a, false, bf, (short)0, acc, false, false);
        }
        const int cb = t * 16 + half * 8;
        float vv[8];
#pragma unroll
        for (int r = 0; r < 8; ++r) {
            int c = cb + r;
            float v = acc[r] + b2[c];
            if ((c & 63) == 0) v = 0.f;    // AC mask: zero DC coefficient
            vv[r] = v;
        }
        *(float4*)&ars[lm][cb]     = make_float4(vv[0], vv[1], vv[2], vv[3]);
        *(float4*)&ars[lm][cb + 4] = make_float4(vv[4], vv[5], vv[6], vv[7]);
    }
    __syncthreads();

    // --- IDCT stage 1: U[i][s] = sum_j A[i][j] * D[j][s] ---
    for (int ss = 0; ss < 12; ++ss) {
        int o = tid + ss * 256;
        int blk = o >> 6, i = (o >> 3) & 7, s = o & 7;
        int p = blk & 15, color = blk >> 4;
        float acc = 0.f;
#pragma unroll
        for (int j = 0; j < 8; ++j) acc += ars[p][color * 64 + i * 8 + j] * Dm[j][s];
        Tb[blk][i * 8 + s] = acc;
    }
    __syncthreads();
    // --- IDCT stage 2: Z[r][s] = sum_i D[i][r] * U[i][s]; out = Z (== freq_out - img) ---
    {
        const size_t outbase = (size_t)b * 3u * 512u * 512u;
        for (int ss = 0; ss < 12; ++ss) {
            int o = tid + ss * 256;
            int blk = o >> 6, r = (o >> 3) & 7, s = o & 7;
            int p = blk & 15, color = blk >> 4;
            float acc = 0.f;
#pragma unroll
            for (int i = 0; i < 8; ++i) acc += Dm[i][r] * Tb[blk][i * 8 + s];
            out[outbase + ((size_t)color * 512u + (size_t)(ih * 8 + r)) * 512u
                        + (size_t)(iw0 * 8 + p * 8 + s)] = acc;
        }
    }
}

extern "C" void kernel_launch(void* const* d_in, const int* in_sizes, int n_in,
                              void* d_out, int out_size, void* d_ws, size_t ws_size,
                              hipStream_t stream) {
    const float* img  = (const float*)d_in[0];
    const float* bits = (const float*)d_in[1];
    const float* w1   = (const float*)d_in[2];
    const float* b1   = (const float*)d_in[3];
    const float* w2   = (const float*)d_in[4];
    const float* b2   = (const float*)d_in[5];
    float* outp = (float*)d_out;

    const size_t W1P = 16u * 8u * 32u * 16u;   // 65536 f16
    const size_t W2P = 12u * 8u * 32u * 16u;   // 49152 f16
    const size_t need = (W1P + W2P) * 2u;

    if (ws_size >= need) {
        _Float16* w1p = (_Float16*)d_ws;
        _Float16* w2p = (_Float16*)((char*)d_ws + W1P * 2u);
        pack_weights<<<dim3(256), dim3(256), 0, stream>>>(w1, w1p, 16);
        pack_weights<<<dim3(192), dim3(256), 0, stream>>>(w2, w2p, 12);
        freq_encoder_kernel<true><<<dim3(4096), dim3(256), 0, stream>>>(
            img, bits, w1, b1, w2, b2, w1p, w2p, outp);
    } else {
        freq_encoder_kernel<false><<<dim3(4096), dim3(256), 0, stream>>>(
            img, bits, w1, b1, w2, b2, nullptr, nullptr, outp);
    }
}
